// GlobalChannelCovariancePooling_24532853195076
// MI455X (gfx1250) — matrix operand order repaired and checked by reading, not compiled, not run on hardware
//
#include <hip/hip_runtime.h>

// ---------------------------------------------------------------------------
// GlobalChannelCovariancePooling for MI455X (gfx1250, wave32, WMMA)
//
// cov[b] = (X - mu)(X - mu)^T / (N-1),  X = x[b] as [C=256, N=4096] row-major.
//
// FP32-accurate fast path: x = hi + lo (f16 split); hi*hi, hi*lo, lo*hi are
// exact in the f32 WMMA accumulator, only the ~2^-22 lo*lo term is dropped.
// Software-pipelined: global b128 prefetch -> WMMA compute (current LDS buf)
// -> convert/split + ds_store (next LDS buf), one barrier per K chunk.
// ---------------------------------------------------------------------------

typedef __attribute__((ext_vector_type(16))) _Float16 v16h;
typedef __attribute__((ext_vector_type(8)))  float    v8f;

#define BATCH 64
#define DIM   256     // channels (output is DIM x DIM per batch)
#define NTOK  4096    // H*W
#define BM    64      // block tile rows
#define BN    128     // block tile cols
#define KC    32      // K chunk = one WMMA K step
#define NCHUNK (NTOK / KC)
#define PSTR  17      // LDS row stride in dword-pairs (34 halves): bank spread
#define NTHR  256     // 8 waves

union FragU { v16h v; unsigned u[8]; };

static __device__ inline unsigned pack2(_Float16 a, _Float16 b) {
  union { _Float16 h[2]; unsigned u; } z;
  z.h[0] = a; z.h[1] = b;
  return z.u;
}

// ---------------------------------------------------------------------------
// Kernel 1: row means. One 256-thread block per (b, c) row of 4096 floats.
// ---------------------------------------------------------------------------
__global__ __launch_bounds__(NTHR) void mean_kernel(const float* __restrict__ x,
                                                    float* __restrict__ mean) {
  const int row = blockIdx.x;                 // b*DIM + c
  const float* p = x + (size_t)row * NTOK;
  float s = 0.f;
  for (int i = threadIdx.x; i < NTOK; i += NTHR) s += p[i];  // coalesced
  __shared__ float red[NTHR];
  red[threadIdx.x] = s;
  __syncthreads();
  for (int off = NTHR / 2; off > 0; off >>= 1) {
    if (threadIdx.x < off) red[threadIdx.x] += red[threadIdx.x + off];
    __syncthreads();
  }
  if (threadIdx.x == 0) mean[row] = red[0] * (1.0f / NTOK);
}

// ---------------------------------------------------------------------------
// Kernel 2: batched SYRK via split-f16 WMMA, double-buffered pipeline.
// Grid: (DIM/BN, DIM/BM, BATCH). Block: 256 threads = 8 wave32.
// Wave w -> 32x32 output slab: wm = w>>2, wn = w&3.
// ---------------------------------------------------------------------------
__global__ __launch_bounds__(NTHR) void cov_kernel(const float* __restrict__ x,
                                                   const float* __restrict__ mean,
                                                   float* __restrict__ out) {
  // Double-buffered LDS: hi/lo f16 tiles as packed dword pairs.
  // Per buffer: A 64*17 + B 128*17 (x2 for hi/lo) = 6528 dwords; x2 bufs ~51KB.
  __shared__ unsigned AsH[2][BM * PSTR];
  __shared__ unsigned AsL[2][BM * PSTR];
  __shared__ unsigned BsH[2][BN * PSTR];
  __shared__ unsigned BsL[2][BN * PSTR];

  const int b  = blockIdx.z;
  const int m0 = blockIdx.y * BM;
  const int n0 = blockIdx.x * BN;

  const float* xb = x + (size_t)b * DIM * NTOK;

  const int t    = threadIdx.x;
  const int lane = t & 31;
  const int w    = t >> 5;
  const int wm   = w >> 2;           // 0..1
  const int wn   = w & 3;            // 0..3
  const int l    = lane & 15;        // M or N within 16x16 tile
  const int h    = lane >> 4;        // lane half

  // Staging map: 8 float4 per 32-wide K chunk row.
  const int kq = t & 7;              // float4 index within row
  const int sr = t >> 3;             // 0..31 base row

  // Means are reused across all K chunks: hoist.
  float mA[2], mB[4];
#pragma unroll
  for (int i = 0; i < 2; ++i) mA[i] = mean[b * DIM + m0 + sr + 32 * i];
#pragma unroll
  for (int i = 0; i < 4; ++i) mB[i] = mean[b * DIM + n0 + sr + 32 * i];

  v8f acc[2][2] = {};                // f32 16x16 accumulators (4 tiles/wave)

  // ---- helpers as lambdas ----
  auto load_chunk = [&](int k0, float4* ra, float4* rb) {
#pragma unroll
    for (int i = 0; i < 2; ++i) {
      const int row = sr + 32 * i;
      ra[i] = *(const float4*)(xb + (size_t)(m0 + row) * NTOK + k0 + kq * 4);
    }
#pragma unroll
    for (int i = 0; i < 4; ++i) {
      const int row = sr + 32 * i;
      rb[i] = *(const float4*)(xb + (size_t)(n0 + row) * NTOK + k0 + kq * 4);
    }
  };

  auto store_chunk = [&](int s, const float4* ra, const float4* rb) {
#pragma unroll
    for (int i = 0; i < 2; ++i) {
      const int row = sr + 32 * i;
      float f0 = ra[i].x - mA[i], f1 = ra[i].y - mA[i];
      float f2 = ra[i].z - mA[i], f3 = ra[i].w - mA[i];
      _Float16 h0 = (_Float16)f0, h1 = (_Float16)f1;
      _Float16 h2 = (_Float16)f2, h3 = (_Float16)f3;
      _Float16 l0 = (_Float16)(f0 - (float)h0), l1 = (_Float16)(f1 - (float)h1);
      _Float16 l2 = (_Float16)(f2 - (float)h2), l3 = (_Float16)(f3 - (float)h3);
      const int pi = row * PSTR + kq * 2;
      AsH[s][pi]     = pack2(h0, h1);
      AsH[s][pi + 1] = pack2(h2, h3);
      AsL[s][pi]     = pack2(l0, l1);
      AsL[s][pi + 1] = pack2(l2, l3);
    }
#pragma unroll
    for (int i = 0; i < 4; ++i) {
      const int row = sr + 32 * i;
      float f0 = rb[i].x - mB[i], f1 = rb[i].y - mB[i];
      float f2 = rb[i].z - mB[i], f3 = rb[i].w - mB[i];
      _Float16 h0 = (_Float16)f0, h1 = (_Float16)f1;
      _Float16 h2 = (_Float16)f2, h3 = (_Float16)f3;
      _Float16 l0 = (_Float16)(f0 - (float)h0), l1 = (_Float16)(f1 - (float)h1);
      _Float16 l2 = (_Float16)(f2 - (float)h2), l3 = (_Float16)(f3 - (float)h3);
      const int pi = row * PSTR + kq * 2;
      BsH[s][pi]     = pack2(h0, h1);
      BsH[s][pi + 1] = pack2(h2, h3);
      BsL[s][pi]     = pack2(l0, l1);
      BsL[s][pi + 1] = pack2(l2, l3);
    }
  };

  auto compute = [&](int s) {
    FragU aH[2], aL[2], bH[2], bL[2];
    // 16-bit A 16x32 layout: lane(l,h), VGPR v holds K pair:
    //   v<4 -> v + 4h ; v>=4 -> v + 4 + 4h   (dword-pair units)
#pragma unroll
    for (int cm = 0; cm < 2; ++cm) {
      const int row = wm * 32 + cm * 16 + l;
#pragma unroll
      for (int v = 0; v < 8; ++v) {
        const int ko = (v < 4) ? (v + 4 * h) : (v + 4 + 4 * h);
        const int pi = row * PSTR + ko;
        aH[cm].u[v] = AsH[s][pi];
        aL[cm].u[v] = AsL[s][pi];
      }
    }
    // 16-bit B 32x16 layout: VGPR v holds K pair (same as A):
    //   v<4 -> v + 4h ; v>=4 -> v + 4 + 4h
#pragma unroll
    for (int cn = 0; cn < 2; ++cn) {
      const int row = wn * 32 + cn * 16 + l;
#pragma unroll
      for (int v = 0; v < 8; ++v) {
        const int pi = row * PSTR + ((v < 4) ? (v + 4 * h) : (v + 4 + 4 * h));
        bH[cn].u[v] = BsH[s][pi];
        bL[cn].u[v] = BsL[s][pi];
      }
    }
    // 12 WMMAs: hi*hi + hi*lo + lo*hi into shared accumulators.
#pragma unroll
    for (int cm = 0; cm < 2; ++cm) {
#pragma unroll
      for (int cn = 0; cn < 2; ++cn) {
        acc[cm][cn] = __builtin_amdgcn_wmma_f32_16x16x32_f16(
            false, aH[cm].v, false, bH[cn].v, (short)0, acc[cm][cn], false, false);
        acc[cm][cn] = __builtin_amdgcn_wmma_f32_16x16x32_f16(
            false, aH[cm].v, false, bL[cn].v, (short)0, acc[cm][cn], false, false);
        acc[cm][cn] = __builtin_amdgcn_wmma_f32_16x16x32_f16(
            false, aL[cm].v, false, bH[cn].v, (short)0, acc[cm][cn], false, false);
      }
    }
  };

  // ---- prologue: stage chunk 0 into buffer 0 ----
  {
    float4 ra[2], rb[4];
    load_chunk(0, ra, rb);
    store_chunk(0, ra, rb);
  }
  __syncthreads();

  // ---- pipelined main loop: one barrier per chunk ----
  for (int c = 0; c < NCHUNK; ++c) {
    const int cur = c & 1;
    float4 ra[2], rb[4];
    const bool more = (c + 1 < NCHUNK);
    if (more) load_chunk((c + 1) * KC, ra, rb);   // global latency hidden
    compute(cur);                                 // WMMAs from current buffer
    if (more) store_chunk(cur ^ 1, ra, rb);       // fill next buffer
    __syncthreads();
  }

  // ---- epilogue: scale by 1/(N-1), store per C-layout (M = r + 8h, N = l) ----
  const float inv = 1.0f / (float)(NTOK - 1);
  float* ob = out + (size_t)b * DIM * DIM;
#pragma unroll
  for (int cm = 0; cm < 2; ++cm) {
#pragma unroll
    for (int cn = 0; cn < 2; ++cn) {
      const int col = n0 + wn * 32 + cn * 16 + l;
#pragma unroll
      for (int r = 0; r < 8; ++r) {
        const int rowO = m0 + wm * 32 + cm * 16 + r + 8 * h;
        ob[(size_t)rowO * DIM + col] = acc[cm][cn][r] * inv;
      }
    }
  }
}

// ---------------------------------------------------------------------------
extern "C" void kernel_launch(void* const* d_in, const int* in_sizes, int n_in,
                              void* d_out, int out_size, void* d_ws, size_t ws_size,
                              hipStream_t stream) {
  (void)in_sizes; (void)n_in; (void)out_size; (void)ws_size;
  const float* x = (const float*)d_in[0];
  float* out     = (float*)d_out;
  float* meanbuf = (float*)d_ws;              // BATCH*DIM floats = 64 KB

  mean_kernel<<<dim3(BATCH * DIM), NTHR, 0, stream>>>(x, meanbuf);
  cov_kernel<<<dim3(DIM / BN, DIM / BM, BATCH), NTHR, 0, stream>>>(x, meanbuf, out);
}
